// SelfCausalAttention_86414741996248
// MI455X (gfx1250) — matrix operand order repaired and checked by reading, not compiled, not run on hardware
//
#include <hip/hip_runtime.h>
#include <hip/hip_bf16.h>
#include <cstdint>

// ---------- CDNA5 WMMA types ----------
typedef __attribute__((ext_vector_type(16))) __bf16 v16bf;
typedef __attribute__((ext_vector_type(8)))  float  v8f;
typedef __attribute__((ext_vector_type(4)))  int    v4i;

union Frag16 {
  v16bf v;
  unsigned int u[8];
  unsigned short s[16];
};

__device__ __forceinline__ unsigned short f32_to_bf16(float f) {
  unsigned int u = __float_as_uint(f);
  u += 0x7FFFu + ((u >> 16) & 1u);          // round-to-nearest-even
  return (unsigned short)(u >> 16);
}
__device__ __forceinline__ unsigned int pack_bf16x2(float a, float b) {
  return (unsigned int)f32_to_bf16(a) | ((unsigned int)f32_to_bf16(b) << 16);
}

// ---------- CDNA5 async global->LDS ----------
#if defined(__has_builtin)
#  if __has_builtin(__builtin_amdgcn_global_load_async_to_lds_b128)
#    define HAVE_ASYNC_LDS 1
#  endif
#endif
#ifndef HAVE_ASYNC_LDS
#  define HAVE_ASYNC_LDS 0
#endif

#define GLOBAL_AS __attribute__((address_space(1)))
#define LDS_AS    __attribute__((address_space(3)))

#if HAVE_ASYNC_LDS
// probe-verified signature: (global int4*, shared int4*, imm offset, imm cpol)
__device__ __forceinline__ void async_copy_b128(const void* g, void* l) {
  __builtin_amdgcn_global_load_async_to_lds_b128(
      (GLOBAL_AS v4i*)(uintptr_t)g,
      (LDS_AS v4i*)(unsigned int)(uintptr_t)l, 0, 0);
}
#endif

#define BATCH 2
#define SEQ   2048
#define EMB   2048
#define NHEAD 16
#define HD    128
#define QKVW  (3 * EMB)   // 6144

// =====================================================================
// fp32 -> packed bf16 converter (4 elements / thread)
// =====================================================================
__global__ void cvt_f32_bf16x4(const float* __restrict__ in,
                               unsigned short* __restrict__ out, int nquads) {
  int i = blockIdx.x * blockDim.x + threadIdx.x;
  if (i < nquads) {
    float4 f = ((const float4*)in)[i];
    uint2 p;
    p.x = pack_bf16x2(f.x, f.y);
    p.y = pack_bf16x2(f.z, f.w);
    ((uint2*)out)[i] = p;
  }
}

// =====================================================================
// bf16 GEMM  C[M,N] = A[M,K] @ B[K,N] + bias   (fp32 accumulate/out)
// block = 256 threads (8 waves); block tile 128x128; wave tile 32x64
// DOUBLE-BUFFERED LDS: tile i+1's async A copies + B global loads are
// issued before tile i's WMMAs; ASYNCcnt drained after compute; one
// barrier per K-step.
// A tile: row-major, pitch 40 halves (80B rows, b128-aligned)
// B tile: TRANSPOSED (Bt[n][k]), pitch 36 halves -> dword frag reads
// =====================================================================
__global__ void __launch_bounds__(256)
gemm_bf16_wmma(const unsigned short* __restrict__ A,
               const unsigned short* __restrict__ Bw,
               const float* __restrict__ bias,
               float* __restrict__ C, int M, int N, int K) {
  __shared__ __align__(16) unsigned short As[2][128 * 40];  // 2 x 10240 B
  __shared__ __align__(16) unsigned short Bt[2][128 * 36];  // 2 x  9216 B

  const int tid  = threadIdx.x;
  const int lane = tid & 31;
  const int wv   = tid >> 5;          // 0..7
  const int wr   = wv >> 1;           // 0..3  (M dir)
  const int wc   = wv & 1;            // 0..1  (N dir)
  const int nl   = lane & 15;
  const int hb   = lane >> 4;

  const int m0 = blockIdx.y * 128;
  const int n0 = blockIdx.x * 128;

  // staging coordinates (fixed per thread)
  const int rA = tid >> 2, cA = tid & 3;    // A: rows rA, rA+64; uint4-col cA
  const int rB = tid >> 4, cB = tid & 15;   // B: rows rB, rB+16; uint4-col cB

  uint4 breg[2];
#if !HAVE_ASYNC_LDS
  uint4 areg[2];
#endif

  auto issue_stage = [&](int k0, int buf) {
#if HAVE_ASYNC_LDS
    async_copy_b128(A + (size_t)(m0 + rA) * K + k0 + cA * 8,
                    &As[buf][rA * 40 + cA * 8]);
    async_copy_b128(A + (size_t)(m0 + rA + 64) * K + k0 + cA * 8,
                    &As[buf][(rA + 64) * 40 + cA * 8]);
#else
    areg[0] = *(const uint4*)(A + (size_t)(m0 + rA) * K + k0 + cA * 8);
    areg[1] = *(const uint4*)(A + (size_t)(m0 + rA + 64) * K + k0 + cA * 8);
#endif
    breg[0] = *(const uint4*)(Bw + (size_t)(k0 + rB) * N + n0 + cB * 8);
    breg[1] = *(const uint4*)(Bw + (size_t)(k0 + rB + 16) * N + n0 + cB * 8);
  };

  auto publish_stage = [&](int buf) {
#pragma unroll
    for (int it = 0; it < 2; ++it) {
      int rb = rB + it * 16;
      const unsigned int* bw32 = (const unsigned int*)&breg[it];
#pragma unroll
      for (int j = 0; j < 8; ++j) {
        unsigned short hv = (unsigned short)(bw32[j >> 1] >> ((j & 1) * 16));
        Bt[buf][(cB * 8 + j) * 36 + rb] = hv;
      }
    }
#if HAVE_ASYNC_LDS
    asm volatile("s_wait_asynccnt 0x0" ::: "memory");
#else
    *(uint4*)&As[buf][rA * 40 + cA * 8] = areg[0];
    *(uint4*)&As[buf][(rA + 64) * 40 + cA * 8] = areg[1];
#endif
  };

  v8f acc[2][4];
  v8f vzero = {};
#pragma unroll
  for (int mi = 0; mi < 2; ++mi)
#pragma unroll
    for (int ni = 0; ni < 4; ++ni) acc[mi][ni] = vzero;

  // ---- prologue: stage tile 0 into buffer 0 ----
  issue_stage(0, 0);
  publish_stage(0);
  __syncthreads();

  for (int k0 = 0; k0 < K; k0 += 32) {
    const int buf = (k0 >> 5) & 1;
    const int nxt = buf ^ 1;
    const bool more = (k0 + 32) < K;
    if (more) issue_stage(k0 + 32, nxt);   // overlap with WMMAs below

    // ---- A fragments (row = lane&15, kb = lane<16?0:8) ----
    Frag16 af[2];
#pragma unroll
    for (int mi = 0; mi < 2; ++mi) {
      int rloc = wr * 32 + mi * 16 + nl;
#pragma unroll
      for (int v = 0; v < 8; ++v) {
        int k = ((lane < 16) ? 0 : 8) + ((v >> 2) << 4) + ((v & 3) << 1);
        af[mi].u[v] = *(const unsigned int*)&As[buf][rloc * 40 + k];
      }
    }
    // ---- B fragments from transposed tile: one dword per VGPR ----
    Frag16 bfr[4];
#pragma unroll
    for (int ni = 0; ni < 4; ++ni) {
      int ncol = wc * 64 + ni * 16 + nl;
#pragma unroll
      for (int v = 0; v < 8; ++v) {
        int kk = ((lane < 16) ? 0 : 8) + ((v >> 2) << 4) + ((v & 3) << 1);
        bfr[ni].u[v] = *(const unsigned int*)&Bt[buf][ncol * 36 + kk];
      }
    }
#pragma unroll
    for (int mi = 0; mi < 2; ++mi)
#pragma unroll
      for (int ni = 0; ni < 4; ++ni)
        acc[mi][ni] = __builtin_amdgcn_wmma_f32_16x16x32_bf16(
            false, af[mi].v, false, bfr[ni].v, (short)0, acc[mi][ni], false, false);

    if (more) publish_stage(nxt);
    __syncthreads();
  }

  // ---- store (C layout: row = v + 8*(lane>=16), col = lane&15) ----
#pragma unroll
  for (int mi = 0; mi < 2; ++mi)
#pragma unroll
    for (int ni = 0; ni < 4; ++ni) {
      int col = n0 + wc * 64 + ni * 16 + nl;
      float bv = bias[col];
#pragma unroll
      for (int v = 0; v < 8; ++v) {
        int row = m0 + wr * 32 + mi * 16 + v + 8 * hb;
        C[(size_t)row * N + col] = acc[mi][ni][v] + bv;
      }
    }
}

// =====================================================================
// Flash causal attention over qkv (fp32, stride 6144), bf16 output.
// grid = (SEQ/128, BATCH*NHEAD); block = 256 (8 waves, 16 q-rows each)
// K tile row-major (pitch 132 halves, b64-aligned rows);
// V tile TRANSPOSED (Vt[hd][key], pitch 66) -> dword frag reads.
// RoPE angle depends on HEAD index (reference quirk); E^-0.5 applied
// AFTER softmax.
// =====================================================================
__global__ void __launch_bounds__(256)
flash_attn_wmma(const float* __restrict__ qkv, unsigned short* __restrict__ obf) {
  __shared__ __align__(16) unsigned short Klds[64 * 132];  // 16896 B
  __shared__ __align__(16) unsigned short Vt[128 * 66];    // 16896 B
  __shared__ __align__(16) unsigned short Plds[8 * 16 * 66];

  const int tid  = threadIdx.x;
  const int lane = tid & 31;
  const int wv   = tid >> 5;
  const int nl   = lane & 15;
  const int hb   = lane >> 4;

  const int q0 = blockIdx.x * 128;
  const int bh = blockIdx.y;
  const int b  = bh >> 4;
  const int h  = bh & 15;
  const float hf = (float)h;

  // ---- Q fragments in registers, RoPE applied at load ----
  Frag16 qf[4];
  {
    int row = q0 + 16 * wv + nl;
    size_t qbase = ((size_t)(b * SEQ + row)) * QKVW + h * HD;
#pragma unroll
    for (int ks = 0; ks < 4; ++ks)
#pragma unroll
      for (int v = 0; v < 8; ++v) {
        int d = ks * 32 + ((lane < 16) ? 0 : 8) + ((v >> 2) << 4) + ((v & 3) << 1);
        float a = qkv[qbase + d], bq = qkv[qbase + d + 1];
        float ang = hf * __expf(-9.2103403719761836f * ((float)(d >> 1) * (1.0f / 64.0f)));
        float sn, cs; __sincosf(ang, &sn, &cs);
        qf[ks].u[v] = pack_bf16x2(a * cs - bq * sn, a * sn + bq * cs);
      }
  }

  // ---- staging trig: this thread always handles pairs 2*(tid&31), +1 ----
  const int cq = tid & 31;       // float4 column within row
  const int rbase = tid >> 5;    // key-row base (stride 8)
  float s0, c0, s1, c1;
  {
    float a0 = hf * __expf(-9.2103403719761836f * ((float)(2 * cq) * (1.0f / 64.0f)));
    float a1 = hf * __expf(-9.2103403719761836f * ((float)(2 * cq + 1) * (1.0f / 64.0f)));
    __sincosf(a0, &s0, &c0);
    __sincosf(a1, &s1, &c1);
  }

  float mrow[8], ssum[8];
  v8f O[8];
  v8f vzero = {};
#pragma unroll
  for (int v = 0; v < 8; ++v) { mrow[v] = -3.0e38f; ssum[v] = 0.0f; }
#pragma unroll
  for (int ot = 0; ot < 8; ++ot) O[ot] = vzero;

  for (int kb0 = 0; kb0 <= q0 + 64; kb0 += 64) {
    // prefetch next key/value tile rows (WGP-near)
    if (kb0 + 64 <= q0 + 64) {
      size_t nb = ((size_t)(b * SEQ + kb0 + 64 + rbase)) * QKVW + h * HD + cq * 4;
      __builtin_prefetch(qkv + nb + EMB, 0, 3);
      __builtin_prefetch(qkv + nb + 2 * EMB, 0, 3);
    }
    // ---- stage K (RoPE) row-major + V transposed, batched float4 ----
#pragma unroll
    for (int half = 0; half < 2; ++half) {
      float4 kreg[4], vreg[4];
#pragma unroll
      for (int it = 0; it < 4; ++it) {
        int rk = rbase + 8 * (half * 4 + it);
        size_t rowb = ((size_t)(b * SEQ + kb0 + rk)) * QKVW + h * HD + cq * 4;
        kreg[it] = *(const float4*)(qkv + rowb + EMB);
        vreg[it] = *(const float4*)(qkv + rowb + 2 * EMB);
      }
#pragma unroll
      for (int it = 0; it < 4; ++it) {
        int rk = rbase + 8 * (half * 4 + it);
        int d = cq * 4;
        float ra = kreg[it].x * c0 - kreg[it].y * s0;
        float rb = kreg[it].x * s0 + kreg[it].y * c0;
        float rc = kreg[it].z * c1 - kreg[it].w * s1;
        float rd = kreg[it].z * s1 + kreg[it].w * c1;
        uint2 pk;
        pk.x = pack_bf16x2(ra, rb);
        pk.y = pack_bf16x2(rc, rd);
        *(uint2*)&Klds[rk * 132 + d] = pk;
        Vt[(d + 0) * 66 + rk] = f32_to_bf16(vreg[it].x);
        Vt[(d + 1) * 66 + rk] = f32_to_bf16(vreg[it].y);
        Vt[(d + 2) * 66 + rk] = f32_to_bf16(vreg[it].z);
        Vt[(d + 3) * 66 + rk] = f32_to_bf16(vreg[it].w);
      }
    }
    __syncthreads();

    // ---- S = Q . K^T   (16x64 per wave, K-dim = hd = 128) ----
    v8f S[4];
#pragma unroll
    for (int ct = 0; ct < 4; ++ct) S[ct] = vzero;
#pragma unroll
    for (int ks = 0; ks < 4; ++ks) {
#pragma unroll
      for (int ct = 0; ct < 4; ++ct) {
        Frag16 kf;
#pragma unroll
        for (int v = 0; v < 8; ++v) {
          int d = ks * 32 + ((lane < 16) ? 0 : 8) + ((v >> 2) << 4) + ((v & 3) << 1);
          kf.u[v] = *(const unsigned int*)&Klds[(ct * 16 + nl) * 132 + d];
        }
        S[ct] = __builtin_amdgcn_wmma_f32_16x16x32_bf16(
            false, qf[ks].v, false, kf.v, (short)0, S[ct], false, false);
      }
    }

    // ---- causal mask + online softmax ----
    const float NEG = -3.0e38f;
#pragma unroll
    for (int ct = 0; ct < 4; ++ct) {
      int keyg = kb0 + ct * 16 + nl;
#pragma unroll
      for (int v = 0; v < 8; ++v) {
        int rowg = q0 + 16 * wv + v + 8 * hb;
        if (keyg > rowg) S[ct][v] = NEG;
      }
    }
    float scl[8];
#pragma unroll
    for (int v = 0; v < 8; ++v) {
      float m = fmaxf(fmaxf(S[0][v], S[1][v]), fmaxf(S[2][v], S[3][v]));
#pragma unroll
      for (int xm = 1; xm < 16; xm <<= 1) m = fmaxf(m, __shfl_xor(m, xm, 32));
      float Mv = fmaxf(mrow[v], m);
      scl[v] = __expf(mrow[v] - Mv);
      mrow[v] = Mv;
    }
#pragma unroll
    for (int ct = 0; ct < 4; ++ct)
#pragma unroll
      for (int v = 0; v < 8; ++v) S[ct][v] = __expf(S[ct][v] - mrow[v]);
#pragma unroll
    for (int v = 0; v < 8; ++v) {
      float rs = S[0][v] + S[1][v] + S[2][v] + S[3][v];
#pragma unroll
      for (int xm = 1; xm < 16; xm <<= 1) rs += __shfl_xor(rs, xm, 32);
      ssum[v] = ssum[v] * scl[v] + rs;
    }
#pragma unroll
    for (int ot = 0; ot < 8; ++ot)
#pragma unroll
      for (int v = 0; v < 8; ++v) O[ot][v] *= scl[v];

    // ---- P -> LDS (C layout -> A layout relayout), same-wave region ----
#pragma unroll
    for (int ct = 0; ct < 4; ++ct)
#pragma unroll
      for (int v = 0; v < 8; ++v)
        Plds[wv * 1056 + (v + 8 * hb) * 66 + ct * 16 + nl] = f32_to_bf16(S[ct][v]);

    // ---- O += P . V   (K-dim = 64 keys), V frags = dword reads ----
#pragma unroll
    for (int ks2 = 0; ks2 < 2; ++ks2) {
      Frag16 pa;
#pragma unroll
      for (int v = 0; v < 8; ++v) {
        int k = ks2 * 32 + ((lane < 16) ? 0 : 8) + ((v >> 2) << 4) + ((v & 3) << 1);
        pa.u[v] = *(const unsigned int*)&Plds[wv * 1056 + nl * 66 + k];
      }
#pragma unroll
      for (int ot = 0; ot < 8; ++ot) {
        Frag16 vb;
#pragma unroll
        for (int v = 0; v < 8; ++v) {
          int kk = ks2 * 32 + ((lane < 16) ? 0 : 8) + ((v >> 2) << 4) + ((v & 3) << 1);
          vb.u[v] = *(const unsigned int*)&Vt[(ot * 16 + nl) * 66 + kk];
        }
        O[ot] = __builtin_amdgcn_wmma_f32_16x16x32_bf16(
            false, pa.v, false, vb.v, (short)0, O[ot], false, false);
      }
    }
    __syncthreads();
  }

  // ---- normalize, scale by E^-0.5 (post-softmax), store bf16 ----
  const float OSCALE = 0.02209708691207961f;  // 2048^-0.5
#pragma unroll
  for (int ot = 0; ot < 8; ++ot)
#pragma unroll
    for (int v = 0; v < 8; ++v) {
      int rowg = q0 + 16 * wv + v + 8 * hb;
      float val = O[ot][v] / ssum[v] * OSCALE;
      obf[((size_t)(b * SEQ + rowg)) * EMB + h * HD + ot * 16 + nl] = f32_to_bf16(val);
    }
}

// =====================================================================
// Launch: cvt -> QKV GEMM -> flash attention -> proj GEMM
// workspace use ~168 MB
// =====================================================================
extern "C" void kernel_launch(void* const* d_in, const int* in_sizes, int n_in,
                              void* d_out, int out_size, void* d_ws, size_t ws_size,
                              hipStream_t stream) {
  const float* x      = (const float*)d_in[0];  // (2,2048,2048)
  const float* w_attn = (const float*)d_in[1];  // (2048,6144)
  const float* b_attn = (const float*)d_in[2];  // (6144)
  const float* w_proj = (const float*)d_in[3];  // (2048,2048)
  const float* b_proj = (const float*)d_in[4];  // (2048)
  float* out = (float*)d_out;

  const int M = BATCH * SEQ;                    // 4096
  char* ws = (char*)d_ws;
  size_t off = 0;
  auto carve = [&](size_t bytes) {
    char* p = ws + off;
    off = (off + bytes + 255) & ~(size_t)255;
    return p;
  };
  unsigned short* xbf  = (unsigned short*)carve((size_t)M * EMB * 2);
  unsigned short* wabf = (unsigned short*)carve((size_t)EMB * QKVW * 2);
  unsigned short* wpbf = (unsigned short*)carve((size_t)EMB * EMB * 2);
  float*          qkv  = (float*)carve((size_t)M * QKVW * 4);
  unsigned short* obf  = (unsigned short*)carve((size_t)M * EMB * 2);

  // fp32 -> bf16 operand packing
  {
    int nq;
    nq = (M * EMB) / 4;
    cvt_f32_bf16x4<<<(nq + 255) / 256, 256, 0, stream>>>(x, xbf, nq);
    nq = (EMB * QKVW) / 4;
    cvt_f32_bf16x4<<<(nq + 255) / 256, 256, 0, stream>>>(w_attn, wabf, nq);
    nq = (EMB * EMB) / 4;
    cvt_f32_bf16x4<<<(nq + 255) / 256, 256, 0, stream>>>(w_proj, wpbf, nq);
  }

  // qkv = x @ w_attn + b_attn
  gemm_bf16_wmma<<<dim3(QKVW / 128, M / 128), 256, 0, stream>>>(
      xbf, wabf, b_attn, qkv, M, QKVW, EMB);

  // flash causal attention with head-indexed RoPE
  flash_attn_wmma<<<dim3(SEQ / 128, BATCH * NHEAD), 256, 0, stream>>>(qkv, obf);

  // out = o @ w_proj + b_proj
  gemm_bf16_wmma<<<dim3(EMB / 128, M / 128), 256, 0, stream>>>(
      obf, wpbf, b_proj, out, M, EMB, EMB);
}
